// LogLinearModel_45105746542758
// MI455X (gfx1250) — matrix operand order repaired and checked by reading, not compiled, not run on hardware
//
#include <hip/hip_runtime.h>

typedef __attribute__((ext_vector_type(16))) _Float16 v16h;
typedef __attribute__((ext_vector_type(8)))  float    v8f;

#define NSYM  64
#define DFEAT 256   // NSYM * (N-1)
#define LROW  4096
#define BROW  32
#define NPAIR 4096  // 64*64 pair rows per table

// ---------------------------------------------------------------------------
// Kernel 1: build pair tables with f16 hi/lo split WMMA (f32-accurate).
//   P01[s0*64+s1][v] = bias[v] + W[v][4*s0+0] + W[v][4*s1+1]
//   P23[s2*64+s3][v] =           W[v][4*s2+2] + W[v][4*s3+3]
// A is a two-hot 16x32 f16 fragment built in registers; B = W^T staged in LDS
// as hi/lo f16 in fragment-contiguous order (one 32B LDS read per lane).
// ---------------------------------------------------------------------------
__global__ void __launch_bounds__(128)
pair_table_wmma(const float* __restrict__ W, const float* __restrict__ bias,
                float* __restrict__ P) {
  __shared__ _Float16 sBhi[DFEAT * NSYM];   // 32 KB
  __shared__ _Float16 sBlo[DFEAT * NSYM];   // 32 KB
  const int tid = threadIdx.x;

  // Stage B[k][n] = W[n][k] as f16 hi/lo, fragment-contiguous:
  // offset o = ((kc*4+vt)*32 + lane)*16 + e ;  k = kc*32 + (lane>>4)*16 + e,
  // n = vt*16 + (lane&15).
  for (int o = tid; o < DFEAT * NSYM; o += 128) {
    int block = o >> 9;
    int lane  = (o >> 4) & 31;
    int e     = o & 15;
    int kc = block >> 2, vt = block & 3;
    int k = kc * 32 + (lane >> 4) * 16 + e;
    int v = vt * 16 + (lane & 15);
    float w = W[v * DFEAT + k];
    _Float16 hi = (_Float16)w;
    _Float16 lo = (_Float16)(w - (float)hi);
    sBhi[o] = hi;
    sBlo[o] = lo;
  }
  __syncthreads();

  const int wave  = tid >> 5;
  const int lane  = tid & 31;
  const int half  = lane >> 4;
  const int nlane = lane & 15;

  for (int it = 0; it < 8; ++it) {
    int gt    = (blockIdx.x * 4 + wave) * 8 + it;  // 0..511 row-tiles
    int table = gt >> 8;                           // 0: P01(+bias), 1: P23
    int tile  = gt & 255;                          // 16-row tile of 4096 rows
    int row   = tile * 16 + nlane;                 // A-matrix row M = lane&15
    int s = row >> 6, t = row & 63;
    int fa = s * 4 + 2 * table;                    // hot feature (slot 2T)
    int fb = t * 4 + 2 * table + 1;                // hot feature (slot 2T+1)
    float* Pbase = P + (size_t)table * (NPAIR * NSYM);

#pragma unroll
    for (int vt = 0; vt < 4; ++vt) {
      float cinit = (table == 0) ? bias[vt * 16 + nlane] : 0.0f;
      v8f c;
#pragma unroll
      for (int r = 0; r < 8; ++r) c[r] = cinit;

#pragma unroll
      for (int kc = 0; kc < 8; ++kc) {
        // Two-hot A fragment: K = kc*32 + half*16 + e
        v16h a;
#pragma unroll
        for (int e = 0; e < 16; ++e) {
          int K = kc * 32 + half * 16 + e;
          a[e] = (K == fa || K == fb) ? (_Float16)1.0f : (_Float16)0.0f;
        }
        const int fo = ((kc * 4 + vt) * 32 + lane) * 16;
        v16h bh = *(const v16h*)&sBhi[fo];
        v16h bl = *(const v16h*)&sBlo[fo];
        c = __builtin_amdgcn_wmma_f32_16x16x32_f16(false, a, false, bh,
                                                   (short)0, c, false, false);
        c = __builtin_amdgcn_wmma_f32_16x16x32_f16(false, a, false, bl,
                                                   (short)0, c, false, false);
      }
#pragma unroll
      for (int r = 0; r < 8; ++r) {
        int M = r + 8 * half;  // C layout: lanes 0-15 M=r, lanes 16-31 M=8+r
        Pbase[(size_t)(tile * 16 + M) * NSYM + vt * 16 + nlane] = c[r];
      }
    }
  }
}

// ---------------------------------------------------------------------------
// Kernel 2: memory-bound main pass. One wave per 16 positions; lane covers 2 vocab
// entries. Two coalesced 256B reads from the L2-resident tables + 1 add.
// ---------------------------------------------------------------------------
__global__ void __launch_bounds__(256)
ngram_gather(const int* __restrict__ x, const float* __restrict__ bias,
             const float* __restrict__ P, float* __restrict__ out) {
  const int gw    = blockIdx.x * 8 + (threadIdx.x >> 5);  // global wave id
  const int lane  = threadIdx.x & 31;
  const int b     = gw >> 8;          // 256 waves per row
  const int jbase = (gw & 255) * 16;  // 16 positions per wave
  const int* xr = x + b * LROW;
  const float* P01 = P;
  const float* P23 = P + NPAIR * NSYM;
  float* outr = out + (size_t)b * LROW * NSYM;

#pragma unroll 4
  for (int i = 0; i < 16; ++i) {
    int j = jbase + i;
    float2 r;
    if (j >= 4) {
      int s0 = xr[j - 4], s1 = xr[j - 3], s2 = xr[j - 2], s3 = xr[j - 1];
      float2 u = *((const float2*)(P01 + (s0 * 64 + s1) * NSYM) + lane);
      float2 w = *((const float2*)(P23 + (s2 * 64 + s3) * NSYM) + lane);
      r.x = u.x + w.x;
      r.y = u.y + w.y;
    } else {  // positions j<4: zero rows of enc -> bias only
      r.x = bias[2 * lane];
      r.y = bias[2 * lane + 1];
    }
    *((float2*)(outr + (size_t)j * NSYM) + lane) = r;
  }
}

// ---------------------------------------------------------------------------
// Fallback (only if d_ws can't hold the 2MB tables): direct exact gather.
// ---------------------------------------------------------------------------
__global__ void __launch_bounds__(256)
ngram_fallback(const int* __restrict__ x, const float* __restrict__ W,
               const float* __restrict__ bias, float* __restrict__ out) {
  int idx = blockIdx.x * 256 + threadIdx.x;
  if (idx >= BROW * LROW * 32) return;
  int lane = idx & 31;
  int pos  = idx >> 5;
  int b = pos >> 12;
  int j = pos & (LROW - 1);
  int v0 = 2 * lane;
  float r0 = bias[v0], r1 = bias[v0 + 1];
  if (j >= 4) {
    const int* xr = x + b * LROW;
#pragma unroll
    for (int p = 0; p < 4; ++p) {
      int f = xr[j - 4 + p] * 4 + p;
      r0 += W[v0 * DFEAT + f];
      r1 += W[(v0 + 1) * DFEAT + f];
    }
  }
  out[(size_t)pos * NSYM + v0]     = r0;
  out[(size_t)pos * NSYM + v0 + 1] = r1;
}

extern "C" void kernel_launch(void* const* d_in, const int* in_sizes, int n_in,
                              void* d_out, int out_size, void* d_ws, size_t ws_size,
                              hipStream_t stream) {
  const int*   x    = (const int*)d_in[0];
  const float* W    = (const float*)d_in[1];
  const float* bias = (const float*)d_in[2];
  float* out = (float*)d_out;

  const size_t tab_bytes = (size_t)2 * NPAIR * NSYM * sizeof(float);  // 2 MB
  if (ws_size >= tab_bytes) {
    float* P = (float*)d_ws;
    // 512 row-tiles / (4 waves * 8 tiles) = 16 blocks
    pair_table_wmma<<<16, 128, 0, stream>>>(W, bias, P);
    // 8192 waves = 131072 positions / 16
    ngram_gather<<<1024, 256, 0, stream>>>(x, bias, P, out);
  } else {
    ngram_fallback<<<(BROW * LROW * 32 + 255) / 256, 256, 0, stream>>>(
        x, W, bias, out);
  }
}
